// Recurrence_19894288515078
// MI455X (gfx1250) — matrix operand order repaired and checked by reading, not compiled, not run on hardware
//
#include <hip/hip_runtime.h>
#include <stdint.h>

// ---------------------------------------------------------------------------
// CDNA5 / gfx1250 implementation. All matmuls via v_wmma_f32_16x16x32_bf16
// (wave32, 16x16 tiles, f32 accumulate). Activations in LDS bf16, weights
// pre-packed into per-fragment bf16 layout in workspace.
// ---------------------------------------------------------------------------

#define DEVINL __device__ __forceinline__

typedef __attribute__((ext_vector_type(16))) __bf16        v16bf;
typedef __attribute__((ext_vector_type(8)))  float         v8f;
typedef __attribute__((ext_vector_type(8)))  unsigned int  v8u;

constexpr int kT = 64, kN = 1024, kL = 16, kH = 256, kCOND = 64;
constexpr int kHCP = 288;        // HC=257 padded to multiple of 32
constexpr int kGEP = 864;        // 3 * kHCP
constexpr int kXIN = 320;        // COND + H
constexpr int kSTATE = 290;      // 1 + 1 + H + L + L

DEVINL unsigned short f2bf(float f) {
  unsigned int u = __builtin_bit_cast(unsigned int, f);
  u += 0x7FFFu + ((u >> 16) & 1u);          // round-to-nearest-even
  return (unsigned short)(u >> 16);
}
DEVINL float bf2f(unsigned short h) {
  unsigned int u = ((unsigned int)h) << 16;
  return __builtin_bit_cast(float, u);
}
DEVINL float sigm(float x) { return 1.0f / (1.0f + __expf(-x)); }

DEVINL v8f z8() {
  v8f v;
  for (int i = 0; i < 8; ++i) v[i] = 0.f;
  return v;
}
DEVINL v16bf mkfrag(uint4 lo, uint4 hi) {
  v8u u = { lo.x, lo.y, lo.z, lo.w, hi.x, hi.y, hi.z, hi.w };
  return __builtin_bit_cast(v16bf, u);
}
// A fragment: 16x32 bf16, row-major source (LDS or global), stride in elements.
// Per ISA layout: lane m=lane&15, half=lane>>4; elems 0..7 <-> K=half*8+0..7,
// elems 8..15 <-> K=16+half*8+0..7  ->  two contiguous 16B loads.
DEVINL v16bf load_a16(const unsigned short* base, int stride, int k0) {
  int lane = threadIdx.x & 31;
  const unsigned short* p = base + (size_t)(lane & 15) * stride + k0 + ((lane >> 4) << 3);
  uint4 lo = *(const uint4*)p;
  uint4 hi = *(const uint4*)(p + 16);
  return mkfrag(lo, hi);
}
// B fragment from pre-packed weights: [tile_n][tile_k][lane][8xu32], 32B/lane.
DEVINL v16bf load_bp(const unsigned int* pack, int tn, int tk, int KT) {
  const unsigned int* p = pack + ((((size_t)tn * KT) + tk) << 8) + ((threadIdx.x & 31) << 3);
  __builtin_prefetch(p + 256, 0, 1);  // global_prefetch_b8 for next fragment
  uint4 lo = *(const uint4*)p;
  uint4 hi = *(const uint4*)(p + 4);
  return mkfrag(lo, hi);
}
DEVINL v8f wmma_bf(v16bf a, v16bf b, v8f c) {
  return __builtin_amdgcn_wmma_f32_16x16x32_bf16(false, a, false, b, (short)0, c, false, false);
}

// ---------------------------------------------------------------------------
// Prep kernels
// ---------------------------------------------------------------------------
__global__ void k_zero_f32(float* p, size_t n) {
  for (size_t i = (size_t)blockIdx.x * blockDim.x + threadIdx.x; i < n;
       i += (size_t)gridDim.x * blockDim.x) p[i] = 0.f;
}
__global__ void k_zero_u16(unsigned short* p, size_t n) {
  for (size_t i = (size_t)blockIdx.x * blockDim.x + threadIdx.x; i < n;
       i += (size_t)gridDim.x * blockDim.x) p[i] = 0;
}
__global__ void k_embed(const float* __restrict__ emb, const int* __restrict__ lines,
                        unsigned short* __restrict__ Mbf) {
  const size_t total = (size_t)kN * kL * kH;
  for (size_t i = (size_t)blockIdx.x * blockDim.x + threadIdx.x; i < total;
       i += (size_t)gridDim.x * blockDim.x) {
    int h = (int)(i & (kH - 1));
    int l = (int)((i >> 8) & (kL - 1));
    int n = (int)(i >> 12);
    int ln = lines[(size_t)n * kL + l];
    Mbf[i] = f2bf(emb[(size_t)ln * kH + h]);
  }
}
// Pack a [gates*srcG, inDim] row-major weight into B fragments with per-gate
// zero padding to dstG columns and inPad=KT*32 K-padding.
__global__ void k_packw(unsigned int* __restrict__ dst, const float* __restrict__ wsrc,
                        int gates, int srcG, int dstG, int inDim, int NT, int KT) {
  const size_t total = (size_t)NT * KT * 256;
  (void)gates;
  for (size_t i = (size_t)blockIdx.x * blockDim.x + threadIdx.x; i < total;
       i += (size_t)gridDim.x * blockDim.x) {
    int v    = (int)(i & 7);
    int lane = (int)((i >> 3) & 31);
    int tk   = (int)((i >> 8) % (size_t)KT);
    int tn   = (int)(i / ((size_t)256 * KT));
    int s0   = 2 * v;
    int kg   = tk * 32 + 8 * (lane >> 4) + s0 + (s0 >= 8 ? 8 : 0);  // K rows kg, kg+1 (table order)
    int ng0  = tn * 16 + (lane & 15);          // padded out column of this lane
    int g0 = ng0 / dstG, c0 = ng0 % dstG;
    unsigned lo = 0, hi = 0;
    if (c0 < srcG && kg < inDim) lo = f2bf(wsrc[(size_t)(g0 * srcG + c0) * inDim + kg]);
    if (c0 < srcG && kg + 1 < inDim) hi = f2bf(wsrc[(size_t)(g0 * srcG + c0) * inDim + kg + 1]);
    dst[i] = lo | (hi << 16);
  }
}
__global__ void k_packb(float* __restrict__ dst, const float* __restrict__ src,
                        int gates, int srcG, int dstG) {
  int total = gates * dstG;
  for (int i = blockIdx.x * blockDim.x + threadIdx.x; i < total;
       i += gridDim.x * blockDim.x) {
    int g = i / dstG, c = i % dstG;
    dst[i] = (c < srcG) ? src[g * srcG + c] : 0.f;
  }
}

// ---------------------------------------------------------------------------
// Encoder: backward GRU scan + all-starts forward GRU scan, builds K and C.
// One block = 16 samples, 8 waves.
// ---------------------------------------------------------------------------
__global__ __launch_bounds__(256) void k_encoder(
    const unsigned short* __restrict__ Mbf,
    unsigned short* __restrict__ FcA, unsigned short* __restrict__ FcB,
    const unsigned int* __restrict__ Pih, const unsigned int* __restrict__ Phh,
    const float* __restrict__ beI, const float* __restrict__ beH,
    float* __restrict__ Kf, float* __restrict__ Cf) {
  __shared__ unsigned short xb[16 * kH];      // current x tile (bf16)
  __shared__ unsigned short gi[16 * kGEP];    // staged x-gates (fwd scan)
  __shared__ float          hbF[16 * kHCP];   // backward h (f32)
  __shared__ unsigned short hbB[16 * kHCP];   // backward h (bf16 WMMA view)

  const int n0 = blockIdx.x * 16;
  const int tid = threadIdx.x;
  const int wv = tid >> 5;
  const int lane = tid & 31;
  const int colt = lane & 15;
  const int halfp = lane >> 4;

  for (int e = tid; e < 16 * kHCP; e += 256) { hbF[e] = 0.f; hbB[e] = 0; }
  __syncthreads();

  // -------- backward scan: h2 = GRU(seq[L-1-s], h); accumulate K/C --------
  for (int s = 0; s < kL; ++s) {
    for (int e = tid; e < 16 * kH; e += 256) {
      int sm = e >> 8, c = e & 255;
      xb[sm * kH + c] = Mbf[((size_t)(n0 + sm) * kL + (kL - 1 - s)) * kH + c];
    }
    __syncthreads();

    float ac[3][4][8];
    for (int r3 = 0; r3 < 3; ++r3) {
      int t = wv + r3 * 8;
      if (t < 18) {
        v8f ar = z8(), az = z8(), aig = z8(), ahg = z8();
        for (int kk = 0; kk < 8; ++kk) {
          v16bf a = load_a16(xb, kH, kk * 32);
          ar  = wmma_bf(a, load_bp(Pih, t,      kk, 8), ar);
          az  = wmma_bf(a, load_bp(Pih, 18 + t, kk, 8), az);
          aig = wmma_bf(a, load_bp(Pih, 36 + t, kk, 8), aig);
        }
        for (int kk = 0; kk < 9; ++kk) {
          v16bf a = load_a16(hbB, kHCP, kk * 32);
          ar  = wmma_bf(a, load_bp(Phh, t,      kk, 9), ar);
          az  = wmma_bf(a, load_bp(Phh, 18 + t, kk, 9), az);
          ahg = wmma_bf(a, load_bp(Phh, 36 + t, kk, 9), ahg);
        }
        for (int e = 0; e < 8; ++e) {
          ac[r3][0][e] = ar[e]; ac[r3][1][e] = az[e];
          ac[r3][2][e] = aig[e]; ac[r3][3][e] = ahg[e];
        }
      }
    }
    __syncthreads();  // all A-reads of hbB done before any write-back
    for (int r3 = 0; r3 < 3; ++r3) {
      int t = wv + r3 * 8;
      if (t < 18) {
        int col = t * 16 + colt;
        float br  = beI[col] + beH[col];
        float bz  = beI[288 + col] + beH[288 + col];
        float big = beI[576 + col];
        float bhg = beH[576 + col];
        for (int e = 0; e < 8; ++e) {
          int sm = e + halfp * 8;
          float rg = sigm(ac[r3][0][e] + br);
          float zg = sigm(ac[r3][1][e] + bz);
          float ng = tanhf(ac[r3][2][e] + big + rg * (ac[r3][3][e] + bhg));
          float hold = hbF[sm * kHCP + col];
          float h2 = (1.f - zg) * ng + zg * hold;
          hbF[sm * kHCP + col] = h2;
          hbB[sm * kHCP + col] = f2bf(h2);
        }
      }
    }
    __syncthreads();
    // B[s] contributes to K[:, j] for j in [0, L-2-s] and C[:, j+1+s, j]
    int nj = kL - 1 - s;
    for (int e = tid; e < nj * 256; e += 256) {
      int j = e >> 8, sm = (e >> 4) & 15, cc = (e & 15) << 4;
      float* kp = Kf + ((size_t)(n0 + sm) * kL + j) * kH + cc;
      for (int q = 0; q < 16; ++q) kp[q] += hbF[sm * kHCP + cc + q];
    }
    for (int e = tid; e < nj * 16; e += 256) {
      int j = e >> 4, sm = e & 15;
      Cf[((size_t)(n0 + sm) * kL + (j + 1 + s)) * kL + j] = hbF[sm * kHCP + kH];
    }
    __syncthreads();
  }

  // -------- forward scan: carry[i] for every start i (256 rows) --------
  for (int j = 0; j < kL; ++j) {
    const unsigned short* Fcur = (j & 1) ? FcB : FcA;
    unsigned short*       Fnxt = (j & 1) ? FcA : FcB;
    for (int e = tid; e < 16 * kH; e += 256) {
      int sm = e >> 8, c = e & 255;
      xb[sm * kH + c] = Mbf[((size_t)(n0 + sm) * kL + j) * kH + c];
    }
    __syncthreads();
    // gi = x @ Wih^T + b_ih  (shared across all starts i)
    for (int t = wv; t < 54; t += 8) {
      v8f acc = z8();
      for (int kk = 0; kk < 8; ++kk)
        acc = wmma_bf(load_a16(xb, kH, kk * 32), load_bp(Pih, t, kk, 8), acc);
      int col = t * 16 + colt;
      for (int e = 0; e < 8; ++e) {
        int sm = e + halfp * 8;
        gi[sm * kGEP + col] = f2bf(acc[e] + beI[col]);
      }
    }
    __syncthreads();
    // gh over 16 row-tiles (i) x 18 col-triples; update rows i<=j
    for (int u = wv; u < 288; u += 8) {
      int m = u / 18, t = u % 18;
      const unsigned short* arow = Fcur + ((size_t)m * kN + n0) * kHCP;
      v8f ahr = z8(), ahz = z8(), ahg = z8();
      for (int kk = 0; kk < 9; ++kk) {
        v16bf a = load_a16(arow, kHCP, kk * 32);
        ahr = wmma_bf(a, load_bp(Phh, t,      kk, 9), ahr);
        ahz = wmma_bf(a, load_bp(Phh, 18 + t, kk, 9), ahz);
        ahg = wmma_bf(a, load_bp(Phh, 36 + t, kk, 9), ahg);
      }
      int col = t * 16 + colt;
      float bhr = beH[col], bhz = beH[288 + col], bhg = beH[576 + col];
      for (int e = 0; e < 8; ++e) {
        int sm = e + halfp * 8;
        size_t fi = ((size_t)m * kN + n0 + sm) * kHCP + col;
        float hold = bf2f(Fcur[fi]);
        float h2 = hold;
        if (m <= j) {
          float rg = sigm(bf2f(gi[sm * kGEP + col]) + ahr[e] + bhr);
          float zg = sigm(bf2f(gi[sm * kGEP + 288 + col]) + ahz[e] + bhz);
          float ng = tanhf(bf2f(gi[sm * kGEP + 576 + col]) + rg * (ahg[e] + bhg));
          h2 = (1.f - zg) * ng + zg * hold;
        }
        Fnxt[fi] = f2bf(h2);
      }
    }
    __threadfence();
    __syncthreads();
    // K[:, j] += sum_{i<=j} carry[i][:H];  C[:, i, j] = carry[i][H]
    {
      int sm = tid >> 4, cc = (tid & 15) << 4;
      float accv[16];
      for (int q = 0; q < 16; ++q) accv[q] = 0.f;
      for (int i = 0; i <= j; ++i) {
        const unsigned short* fr = Fnxt + ((size_t)i * kN + n0 + sm) * kHCP + cc;
        for (int q = 0; q < 16; ++q) accv[q] += bf2f(fr[q]);
      }
      float* kp = Kf + ((size_t)(n0 + sm) * kL + j) * kH + cc;
      for (int q = 0; q < 16; ++q) kp[q] += accv[q];
    }
    for (int e = tid; e < (j + 1) * 16; e += 256) {
      int i = e >> 4, sm = e & 15;
      Cf[((size_t)(n0 + sm) * kL + i) * kL + j] =
          bf2f(Fnxt[((size_t)i * kN + n0 + sm) * kHCP + kH]);
    }
    __syncthreads();
  }
}

// ---------------------------------------------------------------------------
// Decoder: T=64 sequential steps, one block = 16 independent samples.
// ---------------------------------------------------------------------------
__global__ __launch_bounds__(256) void k_decoder(
    const float* __restrict__ cond, const int* __restrict__ actions,
    const float* __restrict__ rnn_hxs, const unsigned short* __restrict__ Mbf,
    const float* __restrict__ Kf, const float* __restrict__ Cf,
    const unsigned int* __restrict__ PW1, const unsigned int* __restrict__ PW2,
    const unsigned int* __restrict__ Pih, const unsigned int* __restrict__ Phh,
    const unsigned int* __restrict__ PWa,
    const float* __restrict__ b1p, const float* __restrict__ b2p,
    const float* __restrict__ bcI, const float* __restrict__ bcH,
    const float* __restrict__ bap, const float* __restrict__ Wc,
    const float* __restrict__ bcv, float* __restrict__ out) {
  __shared__ float          hbF[16 * kH];
  __shared__ unsigned short hbB[16 * kH];
  __shared__ unsigned short xb[16 * kXIN];
  __shared__ unsigned short x1[16 * kH];
  __shared__ float          kS[16 * kH];
  __shared__ float          pS[16 * kL];
  __shared__ float          wS[16 * kL];

  const int n0 = blockIdx.x * 16;
  const int tid = threadIdx.x;
  const int wv = tid >> 5;
  const int lane = tid & 31;
  const int colt = lane & 15;
  const int halfp = lane >> 4;

  if (tid < 16) {
    int s = tid;
    const float* rh = rnn_hxs + (size_t)(n0 + s) * kSTATE;
    bool allz = true;
    for (int c = 0; c < kSTATE; ++c) allz = allz && (rh[c] == 0.f);
    for (int c = 0; c < kH; ++c) hbF[s * kH + c] = rh[2 + c];
    for (int l = 0; l < kL; ++l) pS[s * kL + l] = rh[274 + l];
    if (allz) pS[s * kL + 0] = 1.f;
  }
  __syncthreads();
  for (int e = tid; e < 16 * kH; e += 256) hbB[e] = f2bf(hbF[e]);
  __syncthreads();

  for (int t = 0; t < kT; ++t) {
    // x = [cond_t, p @ M]
    for (int e = tid; e < 16 * kCOND; e += 256) {
      int sm = e >> 6, c = e & 63;
      xb[sm * kXIN + c] = f2bf(cond[((size_t)t * kN + n0 + sm) * kCOND + c]);
    }
    {
      int sm = tid >> 4, d0 = (tid & 15) << 4;
      for (int q = 0; q < 16; ++q) {
        float r = 0.f;
        for (int l = 0; l < kL; ++l)
          r += pS[sm * kL + l] * bf2f(Mbf[((size_t)(n0 + sm) * kL + l) * kH + d0 + q]);
        xb[sm * kXIN + kCOND + d0 + q] = f2bf(r);
      }
    }
    __syncthreads();
    // x1 = relu(x @ W1^T + b1)
    for (int tt = wv; tt < 16; tt += 8) {
      v8f acc = z8();
      for (int kk = 0; kk < 10; ++kk)
        acc = wmma_bf(load_a16(xb, kXIN, kk * 32), load_bp(PW1, tt, kk, 10), acc);
      int col = tt * 16 + colt;
      for (int e = 0; e < 8; ++e) {
        int sm = e + halfp * 8;
        float vx = acc[e] + b1p[col];
        x1[sm * kH + col] = f2bf(vx > 0.f ? vx : 0.f);
      }
    }
    __syncthreads();
    // x2 = relu(x1 @ W2^T + b2) -> xb[:, 0:256]
    for (int tt = wv; tt < 16; tt += 8) {
      v8f acc = z8();
      for (int kk = 0; kk < 8; ++kk)
        acc = wmma_bf(load_a16(x1, kH, kk * 32), load_bp(PW2, tt, kk, 8), acc);
      int col = tt * 16 + colt;
      for (int e = 0; e < 8; ++e) {
        int sm = e + halfp * 8;
        float vx = acc[e] + b2p[col];
        xb[sm * kXIN + col] = f2bf(vx > 0.f ? vx : 0.f);
      }
    }
    __syncthreads();
    // GRU cell: hold all accumulators across the single barrier (RAW on hbB)
    float ac[2][4][8];
    for (int r2 = 0; r2 < 2; ++r2) {
      int tt = wv + r2 * 8;
      v8f ar = z8(), az = z8(), aig = z8(), ahg = z8();
      for (int kk = 0; kk < 8; ++kk) {
        v16bf a = load_a16(xb, kXIN, kk * 32);
        ar  = wmma_bf(a, load_bp(Pih, tt,      kk, 8), ar);
        az  = wmma_bf(a, load_bp(Pih, 16 + tt, kk, 8), az);
        aig = wmma_bf(a, load_bp(Pih, 32 + tt, kk, 8), aig);
      }
      for (int kk = 0; kk < 8; ++kk) {
        v16bf a = load_a16(hbB, kH, kk * 32);
        ar  = wmma_bf(a, load_bp(Phh, tt,      kk, 8), ar);
        az  = wmma_bf(a, load_bp(Phh, 16 + tt, kk, 8), az);
        ahg = wmma_bf(a, load_bp(Phh, 32 + tt, kk, 8), ahg);
      }
      for (int e = 0; e < 8; ++e) {
        ac[r2][0][e] = ar[e]; ac[r2][1][e] = az[e];
        ac[r2][2][e] = aig[e]; ac[r2][3][e] = ahg[e];
      }
    }
    __syncthreads();
    for (int r2 = 0; r2 < 2; ++r2) {
      int tt = wv + r2 * 8;
      int col = tt * 16 + colt;
      float br  = bcI[col] + bcH[col];
      float bz  = bcI[256 + col] + bcH[256 + col];
      float big = bcI[512 + col];
      float bhg = bcH[512 + col];
      for (int e = 0; e < 8; ++e) {
        int sm = e + halfp * 8;
        float rg = sigm(ac[r2][0][e] + br);
        float zg = sigm(ac[r2][1][e] + bz);
        float ng = tanhf(ac[r2][2][e] + big + rg * (ac[r2][3][e] + bhg));
        float hold = hbF[sm * kH + col];
        float h2 = (1.f - zg) * ng + zg * hold;
        hbF[sm * kH + col] = h2;
        hbB[sm * kH + col] = f2bf(h2);
      }
    }
    __syncthreads();
    // k = h @ Wa^T + ba
    for (int tt = wv; tt < 16; tt += 8) {
      v8f acc = z8();
      for (int kk = 0; kk < 8; ++kk)
        acc = wmma_bf(load_a16(hbB, kH, kk * 32), load_bp(PWa, tt, kk, 8), acc);
      int col = tt * 16 + colt;
      for (int e = 0; e < 8; ++e) {
        int sm = e + halfp * 8;
        kS[sm * kH + col] = acc[e] + bap[col];
      }
    }
    __syncthreads();
    // w[n,l] = (K[n,l,:].k[n]) * (p . C[n,:,l])
    {
      int sm = tid >> 4, l = tid & 15;
      const float* kr = Kf + ((size_t)(n0 + sm) * kL + l) * kH;
      float wvv = 0.f;
      for (int d = 0; d < kH; ++d) wvv += kr[d] * kS[sm * kH + d];
      float cv = 0.f;
      for (int i = 0; i < kL; ++i)
        cv += pS[sm * kL + i] * Cf[((size_t)(n0 + sm) * kL + i) * kL + l];
      wS[sm * kL + l] = wvv * cv;
    }
    __syncthreads();
    // write outputs
    for (int e = tid; e < 16 * kH; e += 256) {
      int sm = e >> 8, c = e & 255;
      out[((size_t)t * kN + n0 + sm) * kSTATE + 2 + c] = hbF[sm * kH + c];
    }
    if (tid < 16) {
      int s = tid;
      int a = actions[(size_t)t * kN + n0 + s];
      float mx = -3.4e38f;
      for (int l = 0; l < kL; ++l) mx = fmaxf(mx, wS[s * kL + l]);
      float ex[kL];
      float sum = 0.f;
      for (int l = 0; l < kL; ++l) { ex[l] = __expf(wS[s * kL + l] - mx); sum += ex[l]; }
      float inv = 1.f / sum;
      float vv = bcv[0];
      for (int d = 0; d < kH; ++d) vv += hbF[s * kH + d] * Wc[d];
      float* o = out + ((size_t)t * kN + n0 + s) * kSTATE;
      o[0] = (float)a;
      o[1] = vv;
      for (int l = 0; l < kL; ++l) {
        o[258 + l] = ex[l] * inv;
        float pn = (l == a) ? 1.f : 0.f;
        o[274 + l] = pn;
        pS[s * kL + l] = pn;
      }
    }
    __syncthreads();
  }
  // hx[-1:] tail copy
  __threadfence();
  for (int e = tid; e < 16 * kSTATE; e += 256) {
    int sm = e / kSTATE, c = e % kSTATE;
    out[(size_t)kT * kN * kSTATE + (size_t)(n0 + sm) * kSTATE + c] =
        out[((size_t)(kT - 1) * kN + n0 + sm) * kSTATE + c];
  }
}

// ---------------------------------------------------------------------------
extern "C" void kernel_launch(void* const* d_in, const int* in_sizes, int n_in,
                              void* d_out, int out_size, void* d_ws, size_t ws_size,
                              hipStream_t stream) {
  (void)in_sizes; (void)n_in; (void)out_size; (void)ws_size;
  const float* condition = (const float*)d_in[0];
  const int*   lines     = (const int*)d_in[1];
  const int*   actions   = (const int*)d_in[2];
  const float* rnn_hxs   = (const float*)d_in[3];
  const float* emb       = (const float*)d_in[4];
  const float* w_ih_enc  = (const float*)d_in[5];
  const float* w_hh_enc  = (const float*)d_in[6];
  const float* b_ih_enc  = (const float*)d_in[7];
  const float* b_hh_enc  = (const float*)d_in[8];
  const float* W1        = (const float*)d_in[9];
  const float* b1        = (const float*)d_in[10];
  const float* W2        = (const float*)d_in[11];
  const float* b2        = (const float*)d_in[12];
  const float* w_ih_cell = (const float*)d_in[13];
  const float* w_hh_cell = (const float*)d_in[14];
  const float* b_ih_cell = (const float*)d_in[15];
  const float* b_hh_cell = (const float*)d_in[16];
  const float* Wa        = (const float*)d_in[17];
  const float* ba        = (const float*)d_in[18];
  const float* Wc        = (const float*)d_in[19];
  const float* bc        = (const float*)d_in[20];
  float* out = (float*)d_out;

  char* wsb = (char*)d_ws;
  size_t off = 0;
  auto alloc = [&](size_t bytes) -> void* {
    void* p = wsb + off;
    off = (off + bytes + 255) & ~(size_t)255;
    return p;
  };
  unsigned short* Mbf    = (unsigned short*)alloc((size_t)kN * kL * kH * 2);
  float*          Kf     = (float*)alloc((size_t)kN * kL * kH * 4);
  float*          Cf     = (float*)alloc((size_t)kN * kL * kL * 4);
  unsigned short* FcA    = (unsigned short*)alloc((size_t)kL * kN * kHCP * 2);
  unsigned short* FcB    = (unsigned short*)alloc((size_t)kL * kN * kHCP * 2);
  unsigned int*   PihE   = (unsigned int*)alloc((size_t)54 * 8 * 256 * 4);
  unsigned int*   PhhE   = (unsigned int*)alloc((size_t)54 * 9 * 256 * 4);
  unsigned int*   PihC   = (unsigned int*)alloc((size_t)48 * 8 * 256 * 4);
  unsigned int*   PhhC   = (unsigned int*)alloc((size_t)48 * 8 * 256 * 4);
  unsigned int*   PW1p   = (unsigned int*)alloc((size_t)16 * 10 * 256 * 4);
  unsigned int*   PW2p   = (unsigned int*)alloc((size_t)16 * 8 * 256 * 4);
  unsigned int*   PWap   = (unsigned int*)alloc((size_t)16 * 8 * 256 * 4);
  float* beIp = (float*)alloc(kGEP * 4);
  float* beHp = (float*)alloc(kGEP * 4);
  float* bcIp = (float*)alloc(768 * 4);
  float* bcHp = (float*)alloc(768 * 4);
  float* b1pp = (float*)alloc(256 * 4);
  float* b2pp = (float*)alloc(256 * 4);
  float* bapp = (float*)alloc(256 * 4);

  k_zero_f32<<<1024, 256, 0, stream>>>(Kf, (size_t)kN * kL * kH);
  k_zero_u16<<<1024, 256, 0, stream>>>(FcA, (size_t)kL * kN * kHCP);
  k_zero_u16<<<1024, 256, 0, stream>>>(FcB, (size_t)kL * kN * kHCP);
  k_embed<<<1024, 256, 0, stream>>>(emb, lines, Mbf);

  k_packw<<<256, 256, 0, stream>>>(PihE, w_ih_enc, 3, 257, 288, 256, 54, 8);
  k_packw<<<256, 256, 0, stream>>>(PhhE, w_hh_enc, 3, 257, 288, 257, 54, 9);
  k_packw<<<256, 256, 0, stream>>>(PihC, w_ih_cell, 3, 256, 256, 256, 48, 8);
  k_packw<<<256, 256, 0, stream>>>(PhhC, w_hh_cell, 3, 256, 256, 256, 48, 8);
  k_packw<<<256, 256, 0, stream>>>(PW1p, W1, 1, 256, 256, 320, 16, 10);
  k_packw<<<256, 256, 0, stream>>>(PW2p, W2, 1, 256, 256, 256, 16, 8);
  k_packw<<<256, 256, 0, stream>>>(PWap, Wa, 1, 256, 256, 256, 16, 8);

  k_packb<<<4, 256, 0, stream>>>(beIp, b_ih_enc, 3, 257, 288);
  k_packb<<<4, 256, 0, stream>>>(beHp, b_hh_enc, 3, 257, 288);
  k_packb<<<3, 256, 0, stream>>>(bcIp, b_ih_cell, 3, 256, 256);
  k_packb<<<3, 256, 0, stream>>>(bcHp, b_hh_cell, 3, 256, 256);
  k_packb<<<1, 256, 0, stream>>>(b1pp, b1, 1, 256, 256);
  k_packb<<<1, 256, 0, stream>>>(b2pp, b2, 1, 256, 256);
  k_packb<<<1, 256, 0, stream>>>(bapp, ba, 1, 256, 256);

  k_encoder<<<64, 256, 0, stream>>>(Mbf, FcA, FcB, PihE, PhhE, beIp, beHp, Kf, Cf);
  k_decoder<<<64, 256, 0, stream>>>(condition, actions, rnn_hxs, Mbf, Kf, Cf,
                                    PW1p, PW2p, PihC, PhhC, PWap,
                                    b1pp, b2pp, bcIp, bcHp, bapp, Wc, bc, out);
}
